// WindowAttention3D_64811056496920
// MI455X (gfx1250) — matrix operand order repaired and checked by reading, not compiled, not run on hardware
//
#include <hip/hip_runtime.h>

typedef _Float16 half_t;
typedef __attribute__((ext_vector_type(16))) _Float16 v16h;
typedef __attribute__((ext_vector_type(8)))  _Float16 v8h;
typedef __attribute__((ext_vector_type(8)))  float    v8f;

#define DIM_C   96
#define HEADS   4
#define HDIM    32
#define DISP    4
#define S_TOK   512
#define NTOK    65536   // 2 * 32^3

static __device__ __forceinline__ int laneId() { return (int)(threadIdx.x & 31); }

static __device__ __forceinline__ v8f wmma16(v16h a, v16h b, v8f c) {
  // D = A(16x32 f16) * B(32x16 f16) + C(16x16 f32)
  return __builtin_amdgcn_wmma_f32_16x16x32_f16(false, a, false, b, (short)0, c,
                                                false, false);
}

static __device__ __forceinline__ v8f zero8() {
  v8f z;
#pragma unroll
  for (int i = 0; i < 8; ++i) z[i] = 0.0f;
  return z;
}

// A fragment (16x32 f16). base: row-major, 16 rows, rowStride halfs. kbase: K offset.
// lane l: row = l%16; halves [kbase+c .. +15], c = (l>=16)*16
static __device__ __forceinline__ v16h loadA(const half_t* base, int rowStride, int kbase) {
  int l = laneId();
  const half_t* row = base + (l & 15) * rowStride + kbase + ((l >= 16) ? 16 : 0);
  v8h lo = *(const v8h*)(row);
  v8h hi = *(const v8h*)(row + 8);
  v16h a;
#pragma unroll
  for (int i = 0; i < 8; ++i) { a[i] = lo[i]; a[i + 8] = hi[i]; }
  return a;
}

// B fragment (32x16 f16) where B[k][n] = src[n][k]; src row-major (rows = n).
// lane l: n = l%16; K = kbase + (l>=16)*16 + e -> one contiguous 16-half load
static __device__ __forceinline__ v16h loadB(const half_t* base, int rowStride, int kbase) {
  int l = laneId();
  const half_t* row = base + (l & 15) * rowStride + kbase + ((l >= 16) ? 16 : 0);
  return *(const v16h*)row;
}

// ---------------------------------------------------------------------------
// Pass 1: cyclic-shift gather + QKV projection (M=65536, K=96, N=128 per chunk)
// chunk: 0=Q, 1=K -> [b][h][w][s][d] f16 ; 2=V -> transposed [b][h][w][d][s] f16
// ---------------------------------------------------------------------------
__global__ __launch_bounds__(256) void qkv_kernel(const float* __restrict__ x,
                                                  const float* __restrict__ w_qkv,
                                                  half_t* __restrict__ Qw,
                                                  half_t* __restrict__ Kw,
                                                  half_t* __restrict__ Vt) {
  __shared__ alignas(32) half_t xtile[128 * 96];  // rolled input rows, f16
  __shared__ alignas(32) half_t wT[128 * 96];     // wT[n][k] = w_qkv[k][chunk*128+n]
  const int mblock = blockIdx.x;                  // 512 blocks * 128 rows
  const int chunk  = blockIdx.y;                  // 0,1,2
  const int tid    = threadIdx.x;

  {  // stage transposed weights
    int n = tid >> 1, c0 = (tid & 1) * 48;
#pragma unroll 8
    for (int i = 0; i < 48; ++i)
      wT[n * 96 + c0 + i] = (half_t)w_qkv[(c0 + i) * 384 + chunk * 128 + n];
  }
  {  // stage rolled x rows (roll by -DISP in each spatial axis)
    int r = tid >> 1, c0 = (tid & 1) * 48;
    int mi = mblock * 128 + r;
    int zs = mi & 31, ys = (mi >> 5) & 31, xs = (mi >> 10) & 31, b = mi >> 15;
    int xr = (xs + DISP) & 31, yr = (ys + DISP) & 31, zr = (zs + DISP) & 31;
    const float* src = x + (size_t)(((b * 32 + xr) * 32 + yr) * 32 + zr) * DIM_C;
#pragma unroll 8
    for (int i = 0; i < 48; ++i) xtile[r * 96 + c0 + i] = (half_t)src[c0 + i];
  }
  __syncthreads();

  const int wave = tid >> 5;
  const int l = laneId();
  v8f acc[8];
#pragma unroll
  for (int t = 0; t < 8; ++t) acc[t] = zero8();

#pragma unroll
  for (int ks = 0; ks < 3; ++ks) {
    v16h a = loadA(xtile + wave * 16 * 96, 96, ks * 32);
#pragma unroll
    for (int t = 0; t < 8; ++t) {
      v16h bb = loadB(wT + t * 16 * 96, 96, ks * 32);
      acc[t] = wmma16(a, bb, acc[t]);
    }
  }

  const int lhalf = (l >= 16) ? 8 : 0;
  const int mrowbase = mblock * 128 + wave * 16;
#pragma unroll
  for (int t = 0; t < 8; ++t) {
    int n = t * 16 + (l & 15);         // 0..127 within chunk
    int h = n >> 5, d = n & 31;
#pragma unroll
    for (int r = 0; r < 8; ++r) {
      int mi = mrowbase + r + lhalf;
      int zs = mi & 31, ys = (mi >> 5) & 31, xs = (mi >> 10) & 31, b = mi >> 15;
      int w = (xs >> 3) * 16 + (ys >> 3) * 4 + (zs >> 3);   // window in (x,y,z)
      int s = (xs & 7) * 64 + (ys & 7) * 8 + (zs & 7);      // token in window
      half_t val = (half_t)acc[t][r];
      if (chunk == 2) {
        Vt[((size_t)((b * 4 + h) * 64 + w) * 32 + d) * 512 + s] = val;
      } else {
        half_t* dst = chunk ? Kw : Qw;
        dst[((size_t)((b * 4 + h) * 64 + w) * 512 + s) * 32 + d] = val;
      }
    }
  }
}

// ---------------------------------------------------------------------------
// Pass 2: per-(b,h,window) attention with online softmax.
// Faithful to the reference's relabeling bug: scores/softmax come from window
// qw = (x=(w>>2)&3, y=w&3, z=w>>4) while V and output use window w.
// ---------------------------------------------------------------------------
__global__ __launch_bounds__(256) void attn_kernel(const half_t* __restrict__ Qw,
                                                   const half_t* __restrict__ Kw,
                                                   const half_t* __restrict__ Vt,
                                                   half_t* __restrict__ attnOut) {
  __shared__ alignas(32) half_t Ktile[512 * 32];   // 32 KB
  __shared__ alignas(32) half_t Pst[8 * 16 * 32];  // 8 KB, per-wave P transpose
  const int blk = blockIdx.x;                      // 512 = b*h*w
  const int w = blk & 63, h = (blk >> 6) & 3, b = blk >> 8;
  const int qx = (w >> 2) & 3, qy = w & 3, qz = w >> 4;
  const int qw = qx * 16 + qy * 4 + qz;
  const bool fx = (qx == 3), fy = (qy == 3), fz = (qz == 3);
  const half_t* Qbase = Qw + (size_t)((b * 4 + h) * 64 + qw) * 512 * 32;
  const half_t* Kbase = Kw + (size_t)((b * 4 + h) * 64 + qw) * 512 * 32;
  const half_t* Vbase = Vt + (size_t)((b * 4 + h) * 64 + w) * 32 * 512;
  const int tid = threadIdx.x, wave = tid >> 5, l = tid & 31;
  const int lhalf = (l >= 16) ? 1 : 0;

#pragma unroll
  for (int rr = 0; rr < 2; ++rr) {  // cooperative K stage (2 rows/thread)
    int row = tid * 2 + rr;
    *(v16h*)(Ktile + row * 32)      = *(const v16h*)(Kbase + row * 32);
    *(v16h*)(Ktile + row * 32 + 16) = *(const v16h*)(Kbase + row * 32 + 16);
  }
  __syncthreads();

  const float scale = 0.17677669529663687f;  // 32^-0.5
  half_t* Pw = Pst + wave * 16 * 32;

  for (int it = 0; it < 4; ++it) {
    const int mtile = wave + it * 8;  // 32 m-tiles of 16 query rows
    v16h aq = loadA(Qbase + mtile * 16 * 32, 32, 0);
    v8f O0 = zero8(), O1 = zero8();
    float mrow[8], lrow[8];
    bool ix[8], iy[8], iz[8];
#pragma unroll
    for (int r = 0; r < 8; ++r) {
      mrow[r] = -3.0e38f; lrow[r] = 0.0f;
      int i = mtile * 16 + r + lhalf * 8;
      ix[r] = (i >> 6) >= 4; iy[r] = ((i >> 3) & 7) >= 4; iz[r] = (i & 7) >= 4;
    }

    for (int jc = 0; jc < 16; ++jc) {  // 32 keys per chunk
      const int jb = jc * 32;
      v16h bk0 = loadB(Ktile + jb * 32, 32, 0);
      v16h bk1 = loadB(Ktile + (jb + 16) * 32, 32, 0);
      v8f s0 = wmma16(aq, bk0, zero8());
      v8f s1 = wmma16(aq, bk1, zero8());

      int j0 = jb + (l & 15), j1 = j0 + 16;
      bool j0x = (j0 >> 6) >= 4, j0y = ((j0 >> 3) & 7) >= 4, j0z = (j0 & 7) >= 4;
      bool j1x = (j1 >> 6) >= 4, j1y = ((j1 >> 3) & 7) >= 4, j1z = (j1 & 7) >= 4;

      float cmax[8];
#pragma unroll
      for (int r = 0; r < 8; ++r) {
        float a0 = s0[r] * scale, a1 = s1[r] * scale;
        bool k0 = (fx & (ix[r] != j0x)) | (fy & (iy[r] != j0y)) | (fz & (iz[r] != j0z));
        bool k1 = (fx & (ix[r] != j1x)) | (fy & (iy[r] != j1y)) | (fz & (iz[r] != j1z));
        if (k0) a0 = -1.0e30f;
        if (k1) a1 = -1.0e30f;
        s0[r] = a0; s1[r] = a1;
        float v = fmaxf(a0, a1);
        v = fmaxf(v, __shfl_xor(v, 1, 32));
        v = fmaxf(v, __shfl_xor(v, 2, 32));
        v = fmaxf(v, __shfl_xor(v, 4, 32));
        v = fmaxf(v, __shfl_xor(v, 8, 32));
        cmax[r] = v;
      }
#pragma unroll
      for (int r = 0; r < 8; ++r) {
        float mnew = fmaxf(mrow[r], cmax[r]);
        float corr = __expf(mrow[r] - mnew);
        float p0 = __expf(s0[r] - mnew);
        float p1 = __expf(s1[r] - mnew);
        s0[r] = p0; s1[r] = p1;
        float ps = p0 + p1;
        ps += __shfl_xor(ps, 1, 32);
        ps += __shfl_xor(ps, 2, 32);
        ps += __shfl_xor(ps, 4, 32);
        ps += __shfl_xor(ps, 8, 32);
        lrow[r] = lrow[r] * corr + ps;
        mrow[r] = mnew;
        O0[r] *= corr; O1[r] *= corr;
      }
      // transpose P (C-layout -> A-layout) through per-wave LDS tile
#pragma unroll
      for (int r = 0; r < 8; ++r) {
        int prow = r + lhalf * 8;
        Pw[prow * 32 + (l & 15)]      = (half_t)s0[r];
        Pw[prow * 32 + 16 + (l & 15)] = (half_t)s1[r];
      }
      asm volatile("s_wait_dscnt 0" ::: "memory");
      v16h ap = loadA(Pw, 32, 0);
      const half_t* v0p = Vbase + (size_t)(l & 15) * 512 + jb + ((l >= 16) ? 16 : 0);
      v16h bv0 = *(const v16h*)v0p;
      v16h bv1 = *(const v16h*)(v0p + 16 * 512);
      O0 = wmma16(ap, bv0, O0);
      O1 = wmma16(ap, bv1, O1);
    }

#pragma unroll
    for (int r = 0; r < 8; ++r) {  // normalize + un-window scatter (shifted space)
      int i = mtile * 16 + r + lhalf * 8;
      int sx = i >> 6, sy = (i >> 3) & 7, sz = i & 7;
      int X = (w >> 4) * 8 + sx, Y = ((w >> 2) & 3) * 8 + sy, Z = (w & 3) * 8 + sz;
      size_t tok = (size_t)((b * 32 + X) * 32 + Y) * 32 + Z;
      float inv = 1.0f / lrow[r];
      attnOut[tok * 128 + h * 32 + (l & 15)]      = (half_t)(O0[r] * inv);
      attnOut[tok * 128 + h * 32 + 16 + (l & 15)] = (half_t)(O1[r] * inv);
    }
  }
}

// ---------------------------------------------------------------------------
// Pass 3: output projection (M=65536, K=128, N=96) + bias + roll(+DISP) store
// ---------------------------------------------------------------------------
__global__ __launch_bounds__(256) void proj_kernel(const half_t* __restrict__ attnOut,
                                                   const float* __restrict__ w_out,
                                                   const float* __restrict__ b_out,
                                                   float* __restrict__ out) {
  __shared__ alignas(32) half_t wT[96 * 128];  // wT[n][k] = w_out[k][n]
  __shared__ float sbias[96];
  const int tid = threadIdx.x;
  if (tid < 192) {
    int n = tid >> 1, k0 = (tid & 1) * 64;
#pragma unroll 8
    for (int i = 0; i < 64; ++i) wT[n * 128 + k0 + i] = (half_t)w_out[(k0 + i) * 96 + n];
  }
  if (tid < 96) sbias[tid] = b_out[tid];
  __syncthreads();

  const int wave = tid >> 5, l = tid & 31, lhalf = (l >= 16) ? 8 : 0;
  const int mbase = blockIdx.x * 128 + wave * 16;
  const half_t* Arow = attnOut + (size_t)mbase * 128;

  v8f acc[6];
#pragma unroll
  for (int t = 0; t < 6; ++t) acc[t] = zero8();
#pragma unroll
  for (int ks = 0; ks < 4; ++ks) {
    v16h a = loadA(Arow, 128, ks * 32);
#pragma unroll
    for (int t = 0; t < 6; ++t) {
      v16h bb = loadB(wT + t * 16 * 128, 128, ks * 32);
      acc[t] = wmma16(a, bb, acc[t]);
    }
  }
#pragma unroll
  for (int t = 0; t < 6; ++t) {
    int n = t * 16 + (l & 15);  // 0..95
    float bias = sbias[n];
#pragma unroll
    for (int r = 0; r < 8; ++r) {
      int mi = mbase + r + lhalf;
      int zs = mi & 31, ys = (mi >> 5) & 31, xs = (mi >> 10) & 31, b = mi >> 15;
      int Xo = (xs + DISP) & 31, Yo = (ys + DISP) & 31, Zo = (zs + DISP) & 31;
      out[(size_t)(((b * 32 + Xo) * 32 + Yo) * 32 + Zo) * 96 + n] = acc[t][r] + bias;
    }
  }
}

extern "C" void kernel_launch(void* const* d_in, const int* in_sizes, int n_in,
                              void* d_out, int out_size, void* d_ws, size_t ws_size,
                              hipStream_t stream) {
  (void)in_sizes; (void)n_in; (void)out_size; (void)ws_size;
  const float* x     = (const float*)d_in[0];
  const float* w_qkv = (const float*)d_in[1];
  const float* w_out = (const float*)d_in[2];
  const float* b_out = (const float*)d_in[3];
  float* out = (float*)d_out;

  const size_t nQ = (size_t)2 * HEADS * 64 * S_TOK * HDIM;  // 8,388,608 halfs
  half_t* Qw = (half_t*)d_ws;
  half_t* Kw = Qw + nQ;
  half_t* Vt = Kw + nQ;
  half_t* attnOut = Vt + nQ;  // total workspace: 64 MB

  qkv_kernel<<<dim3(NTOK / 128, 3), 256, 0, stream>>>(x, w_qkv, Qw, Kw, Vt);
  attn_kernel<<<dim3(512), 256, 0, stream>>>(Qw, Kw, Vt, attnOut);
  proj_kernel<<<dim3(512), 256, 0, stream>>>(attnOut, w_out, b_out, out);
}
